// EASeq2SeqLSTM_63410897158897
// MI455X (gfx1250) — compile-verified, matrix operand order checked by reading, not run on hardware
//
#include <hip/hip_runtime.h>
#include <hip/hip_bf16.h>

// EA-LSTM seq2seq on MI455X (gfx1250, wave32).
//
// Strategy: batch rows are independent through the whole recurrence, so
// grid = 16 persistent workgroups x 16 batch rows. Each WG:
//   - 512 threads = 16 waves; wave w owns hidden columns [16w,16w+16) and the
//     four gate WMMA tiles {w, w+16, w+32, w+48} of the [16 x 4H] gate GEMM.
//   - per step: A fragments (activations, bf16) from LDS / global seq buffer,
//     B fragments (weights, bf16, pre-tiled per-lane-contiguous) streamed from
//     L2, v_wmma_f32_16x16x32_bf16 accumulate, elementwise LSTM cell update in
//     registers, h written back to LDS as bf16. Two s_barriers per step.
// Weights (~3.2 MB bf16) are L2-resident and shared by all 16 WGs.
// A compiler memory barrier at the top of each timestep prevents LICM from
// hoisting the (t-invariant) weight loads out of the loop and spilling them
// to per-thread scratch (32 MB L2 footprint vs 3.2 MB shared).

typedef __bf16 bf16_t;
typedef __attribute__((ext_vector_type(8)))  __bf16 v8bf;
typedef __attribute__((ext_vector_type(16))) __bf16 v16bf;
typedef __attribute__((ext_vector_type(8)))  float  v8f;

#define B_SZ   256
#define L_SZ   365
#define F_SZ   32
#define S_SZ   27
#define H_SZ   256
#define HOR_SZ 24

// workspace offsets in bf16 elements
#define OFF_EA  0           // 9*64*512  = 294912  (EA W, K=288,N=1024)
#define OFF_ENC 294912      // 16*64*512 = 524288  (enc [Wih;Whh], K=512,N=1024)
#define OFF_D0  819200      // 8*64*512  = 262144  (dec Whh0, K=256,N=1024)
#define OFF_D1  1081344     // 16*64*512 = 524288  (dec [Wih1;Whh1], K=512,N=1024)
#define OFF_SEQ 1605632     // 256*365*256 bf16 ea hidden sequence
// total ws requirement: (1605632 + 23920640)*2 bytes ~= 51 MB

// Prevents loop-invariant code motion of memory loads across timesteps
// (emits no instruction; blocks the weight-fragment spill-to-scratch pattern).
#define KEEP_LOADS_IN_LOOP() asm volatile("" ::: "memory")

__device__ __forceinline__ float sigm(float x)  { return 1.0f / (1.0f + __expf(-x)); }
__device__ __forceinline__ float tanh_f(float x){ float e = __expf(2.0f * x); return 1.0f - 2.0f / (e + 1.0f); }

// A-operand fragment, 16-bit 16x32 layout:
// lane<16 : M=lane,    K = {kb..kb+7}  and {kb+16..kb+23}
// lane>=16: M=lane-16, K = {kb+8..+15} and {kb+24..+31}
__device__ __forceinline__ v16bf load_a_frag(const bf16_t* rowbase, int kbase, int lane) {
  int hi = (lane & 16) ? 8 : 0;
  v8bf lo = *(const v8bf*)(rowbase + kbase + hi);
  v8bf hh = *(const v8bf*)(rowbase + kbase + 16 + hi);
  return __builtin_shufflevector(lo, hh, 0,1,2,3,4,5,6,7,8,9,10,11,12,13,14,15);
}

__device__ __forceinline__ v16bf load_b_frag(const bf16_t* wl, size_t tile) {
  return *(const v16bf*)(wl + tile * 512);
}

#define WMMA_BF16(A, Bm, C) \
  __builtin_amdgcn_wmma_f32_16x16x32_bf16(false, (A), false, (Bm), (short)0, (C), false, false)

// One K-slice of the four gate tiles: load A + 4 B fragments, then 4 WMMAs.
#define GATE_K_SLICE(AROW, KT_, WPTR, KTILES, a0, a1, a2, a3)                            \
  do {                                                                                   \
    v16bf av = load_a_frag((AROW), (KT_) * 32, lane);                                    \
    const bf16_t* wl = (WPTR) + lane * 16;                                               \
    v16bf b0 = load_b_frag(wl, (size_t)(0 * 16 + wave) * (KTILES) + (KT_));              \
    v16bf b1 = load_b_frag(wl, (size_t)(1 * 16 + wave) * (KTILES) + (KT_));              \
    v16bf b2 = load_b_frag(wl, (size_t)(2 * 16 + wave) * (KTILES) + (KT_));              \
    v16bf b3 = load_b_frag(wl, (size_t)(3 * 16 + wave) * (KTILES) + (KT_));              \
    a0 = WMMA_BF16(av, b0, a0);                                                          \
    a1 = WMMA_BF16(av, b1, a1);                                                          \
    a2 = WMMA_BF16(av, b2, a2);                                                          \
    a3 = WMMA_BF16(av, b3, a3);                                                          \
  } while (0)

// --- prep: convert + tile the four big weight matrices into bf16 B layout ---
// tile ti = nt*KT + kt; element (lane, j) stored at ti*512 + lane*16 + j with
// n = nt*16 + (lane&15), K = kt*32 + j + 8*((j>=8) + (lane>=16)).
__global__ void __launch_bounds__(256)
prep_weights(const float* __restrict__ Wf, const float* __restrict__ Wo,
             const float* __restrict__ Wi, const float* __restrict__ Wg,
             const float* __restrict__ eWih, const float* __restrict__ eWhh,
             const float* __restrict__ dWhh0,
             const float* __restrict__ dWih1, const float* __restrict__ dWhh1,
             bf16_t* __restrict__ ws)
{
  const int EA_N  = 9  * 64 * 512;
  const int ENC_N = 16 * 64 * 512;
  const int D0_N  = 8  * 64 * 512;
  const int D1_N  = 16 * 64 * 512;
  const int total = EA_N + ENC_N + D0_N + D1_N;
  for (int idx = blockIdx.x * blockDim.x + threadIdx.x; idx < total;
       idx += gridDim.x * blockDim.x) {
    int which, e, KT; bf16_t* dst;
    if (idx < EA_N)                    { which = 0; e = idx;                      KT = 9;  dst = ws + OFF_EA;  }
    else if (idx < EA_N + ENC_N)       { which = 1; e = idx - EA_N;               KT = 16; dst = ws + OFF_ENC; }
    else if (idx < EA_N + ENC_N + D0_N){ which = 2; e = idx - EA_N - ENC_N;        KT = 8;  dst = ws + OFF_D0;  }
    else                               { which = 3; e = idx - EA_N - ENC_N - D0_N; KT = 16; dst = ws + OFF_D1; }
    int ti = e >> 9, r = e & 511, lane = r >> 4, j = r & 15;
    int nt = ti / KT, kt = ti % KT;
    int n = nt * 16 + (lane & 15);
    int K = kt * 32 + j + 8 * ((j >= 8) + (lane >= 16));
    float v;
    if (which == 0) {               // EA: columns ordered [f | o | i | g], K over [x(32), h(256)]
      int g = n >> 8, cc = n & 255;
      const float* W = (g == 0) ? Wf : (g == 1) ? Wo : (g == 2) ? Wi : Wg;
      v = W[(size_t)cc * 288 + K];
    } else if (which == 1) {        // enc: K over [ea_h(256), h(256)], torch gate order i,f,g,o
      v = (K < 256) ? eWih[(size_t)n * 256 + K] : eWhh[(size_t)n * 256 + (K - 256)];
    } else if (which == 2) {        // dec layer0 Whh0
      v = dWhh0[(size_t)n * 256 + K];
    } else {                        // dec layer1: K over [hA(256), hB(256)]
      v = (K < 256) ? dWih1[(size_t)n * 256 + K] : dWhh1[(size_t)n * 256 + (K - 256)];
    }
    dst[e] = (bf16_t)v;
  }
}

// ------------------------------- main kernel -------------------------------
__global__ void __launch_bounds__(512, 1)
ea_seq2seq_main(const float* __restrict__ x, const float* __restrict__ s,
                const float* __restrict__ bfv, const float* __restrict__ bov,
                const float* __restrict__ biv, const float* __restrict__ bgv,
                const float* __restrict__ Wsi, const float* __restrict__ bsi,
                const float* __restrict__ Wsg, const float* __restrict__ bsg,
                const float* __restrict__ ebih, const float* __restrict__ ebhh,
                const float* __restrict__ w0col, const float* __restrict__ dbih0,
                const float* __restrict__ dbhh0, const float* __restrict__ dbih1,
                const float* __restrict__ dbhh1, const float* __restrict__ fcW,
                const float* __restrict__ fcb,
                const bf16_t* __restrict__ Wea,  const bf16_t* __restrict__ Wenc,
                const bf16_t* __restrict__ Wd0,  const bf16_t* __restrict__ Wd1,
                bf16_t* __restrict__ seq, float* __restrict__ out)
{
  __shared__ bf16_t xh[16][512];    // activations, row-major, K stride 1
  __shared__ float  part[16][256];  // fc partial products (deterministic reduce)
  __shared__ float  din_s[16];      // decoder feedback input per row

  const int tid   = threadIdx.x;
  const int wave  = tid >> 5;
  const int lane  = tid & 31;
  const int c     = lane & 15;         // column within wave tile / A row index
  const int halfm = (lane >> 4) & 1;   // C/D layout: m = r + 8*halfm
  const int col   = wave * 16 + c;     // owned hidden column
  const int brow0 = blockIdx.x * 16;   // batch-row tile base

  { // zero xh (initial h == 0)
    bf16_t* p = &xh[0][0];
    for (int i = tid; i < 16 * 512; i += 512) p[i] = (bf16_t)0.0f;
  }

  // static contributions si/sg, held in registers per owned (row, col) element
  float si_r[8], sg_r[8];
#pragma unroll
  for (int r = 0; r < 8; ++r) {
    int m = r + 8 * halfm;
    const float* srow = s + (size_t)(brow0 + m) * S_SZ;
    float ai = bsi[col], ag = bsg[col];
    for (int k = 0; k < S_SZ; ++k) {
      float sv = srow[k];
      ai += sv * Wsi[col * S_SZ + k];
      ag += sv * Wsg[col * S_SZ + k];
    }
    si_r[r] = ai; sg_r[r] = ag;
  }
  const float bfr = bfv[col], bor = bov[col], bir = biv[col], bgr = bgv[col];

  float cst[8], hreg[8];
#pragma unroll
  for (int r = 0; r < 8; ++r) { cst[r] = 0.f; hreg[r] = 0.f; }

  __syncthreads();

  const bf16_t* arow = &xh[c][0];   // per-lane A row base in LDS

  // ------------------------------ EA phase ------------------------------
  for (int t = 0; t < L_SZ; ++t) {
    KEEP_LOADS_IN_LOOP();
    { // stage x_t into cols [0,32)
      int rr = tid >> 5, ff = tid & 31;
      xh[rr][ff] = (bf16_t)x[((size_t)(brow0 + rr) * L_SZ + t) * F_SZ + ff];
    }
    __syncthreads();
    v8f a0 = {0.f,0.f,0.f,0.f,0.f,0.f,0.f,0.f}, a1 = a0, a2 = a0, a3 = a0;
    for (int kt = 0; kt < 9; ++kt) {
      GATE_K_SLICE(arow, kt, Wea, 9, a0, a1, a2, a3);
    }
    __syncthreads();   // all reads of xh done before h/x rewrites
#pragma unroll
    for (int r = 0; r < 8; ++r) {
      float fv = sigm(a0[r] + bfr);
      float ov = sigm(a1[r] + bor);
      float iv = sigm(a2[r] + bir + si_r[r]);
      float gv = tanh_f(a3[r] + bgr + sg_r[r]);
      cst[r] = fv * cst[r] + iv * gv;
      float hv = ov * tanh_f(cst[r]);
      int m = r + 8 * halfm;
      bf16_t hb = (bf16_t)hv;
      xh[m][F_SZ + col] = hb;                                   // h for next step
      seq[((size_t)(brow0 + m) * L_SZ + t) * H_SZ + col] = hb;  // ea_seq for encoder
    }
    // ordering vs next-iter matmul is enforced by the barrier after x staging
  }
  __syncthreads();

  // encoder init: h_enc = h_ea (copy cols [32,288) -> [256,512)), c_enc = c_ea (in regs)
  for (int i = tid; i < 16 * 256; i += 512) {
    int m = i >> 8, cc = i & 255;
    xh[m][256 + cc] = xh[m][F_SZ + cc];
  }
  __syncthreads();

  const float be0 = ebih[0 * 256 + col] + ebhh[0 * 256 + col];  // i
  const float be1 = ebih[1 * 256 + col] + ebhh[1 * 256 + col];  // f
  const float be2 = ebih[2 * 256 + col] + ebhh[2 * 256 + col];  // g
  const float be3 = ebih[3 * 256 + col] + ebhh[3 * 256 + col];  // o

  // ---------------------------- encoder phase ----------------------------
  for (int t = 0; t < L_SZ; ++t) {
    KEEP_LOADS_IN_LOOP();
    v8f a0 = {0.f,0.f,0.f,0.f,0.f,0.f,0.f,0.f}, a1 = a0, a2 = a0, a3 = a0;
    const bf16_t* grow = seq + ((size_t)(brow0 + c) * L_SZ + t) * H_SZ;
    for (int kt = 0; kt < 8; ++kt) {       // K 0..255: ea_seq[t] from global (L2)
      GATE_K_SLICE(grow, kt, Wenc, 16, a0, a1, a2, a3);
    }
    for (int kt = 8; kt < 16; ++kt) {      // K 256..511: h_enc from LDS
      GATE_K_SLICE(arow, kt, Wenc, 16, a0, a1, a2, a3);
    }
    __syncthreads();
#pragma unroll
    for (int r = 0; r < 8; ++r) {
      float iv = sigm(a0[r] + be0);
      float fv = sigm(a1[r] + be1);
      float gv = tanh_f(a2[r] + be2);
      float ov = sigm(a3[r] + be3);
      cst[r] = fv * cst[r] + iv * gv;
      hreg[r] = ov * tanh_f(cst[r]);
      int m = r + 8 * halfm;
      xh[m][256 + col] = (bf16_t)hreg[r];
    }
    __syncthreads();
  }

  // ---------------------------- decoder phase ----------------------------
  float cA[8], cB[8];
#pragma unroll
  for (int r = 0; r < 8; ++r) { cA[r] = cst[r]; cB[r] = cst[r]; }
#pragma unroll
  for (int r = 0; r < 8; ++r) {            // hA = h_enc into cols [0,256); hB already at [256,512)
    int m = r + 8 * halfm;
    xh[m][col] = (bf16_t)hreg[r];
  }
  if (tid < 16) din_s[tid] = 0.f;
  __syncthreads();

  const float b00 = dbih0[0*256+col] + dbhh0[0*256+col], b01 = dbih0[1*256+col] + dbhh0[1*256+col];
  const float b02 = dbih0[2*256+col] + dbhh0[2*256+col], b03 = dbih0[3*256+col] + dbhh0[3*256+col];
  const float b10 = dbih1[0*256+col] + dbhh1[0*256+col], b11 = dbih1[1*256+col] + dbhh1[1*256+col];
  const float b12 = dbih1[2*256+col] + dbhh1[2*256+col], b13 = dbih1[3*256+col] + dbhh1[3*256+col];
  const float w00 = w0col[0*256+col], w01 = w0col[1*256+col];
  const float w02 = w0col[2*256+col], w03 = w0col[3*256+col];
  const float fcwr = fcW[col];
  const float fcbv = fcb[0];

  for (int t = 0; t < HOR_SZ; ++t) {
    KEEP_LOADS_IN_LOOP();
    // layer 0: hA @ Whh0^T (K=256) + rank-1 din * Wih0 column
    v8f a0 = {0.f,0.f,0.f,0.f,0.f,0.f,0.f,0.f}, a1 = a0, a2 = a0, a3 = a0;
    for (int kt = 0; kt < 8; ++kt) {
      GATE_K_SLICE(arow, kt, Wd0, 8, a0, a1, a2, a3);
    }
    __syncthreads();
#pragma unroll
    for (int r = 0; r < 8; ++r) {
      int m = r + 8 * halfm;
      float dv = din_s[m];
      float iv = sigm(a0[r] + b00 + dv * w00);
      float fv = sigm(a1[r] + b01 + dv * w01);
      float gv = tanh_f(a2[r] + b02 + dv * w02);
      float ov = sigm(a3[r] + b03 + dv * w03);
      cA[r] = fv * cA[r] + iv * gv;
      float hAv = ov * tanh_f(cA[r]);
      xh[m][col] = (bf16_t)hAv;
    }
    __syncthreads();
    // layer 1: [hA, hB] @ [Wih1; Whh1]^T (K=512)
    a0 = (v8f){0.f,0.f,0.f,0.f,0.f,0.f,0.f,0.f}; a1 = a0; a2 = a0; a3 = a0;
    for (int kt = 0; kt < 16; ++kt) {
      GATE_K_SLICE(arow, kt, Wd1, 16, a0, a1, a2, a3);
    }
    __syncthreads();
#pragma unroll
    for (int r = 0; r < 8; ++r) {
      int m = r + 8 * halfm;
      float iv = sigm(a0[r] + b10);
      float fv = sigm(a1[r] + b11);
      float gv = tanh_f(a2[r] + b12);
      float ov = sigm(a3[r] + b13);
      cB[r] = fv * cB[r] + iv * gv;
      float hBv = ov * tanh_f(cB[r]);
      xh[m][256 + col] = (bf16_t)hBv;
      part[m][wave * 16 + c] = hBv * fcwr;     // fc partial product
    }
    __syncthreads();
    if (tid < 16) {   // deterministic fc reduction + feedback + output
      float p = fcbv;
      for (int i = 0; i < 256; ++i) p += part[tid][i];
      din_s[tid] = p;
      out[(size_t)(brow0 + tid) * HOR_SZ + t] = p;
    }
    __syncthreads();
  }
}

extern "C" void kernel_launch(void* const* d_in, const int* in_sizes, int n_in,
                              void* d_out, int out_size, void* d_ws, size_t ws_size,
                              hipStream_t stream)
{
  const float* x     = (const float*)d_in[0];
  const float* s     = (const float*)d_in[1];
  const float* Wf    = (const float*)d_in[2];
  const float* bf_   = (const float*)d_in[3];
  const float* Wo    = (const float*)d_in[4];
  const float* bo_   = (const float*)d_in[5];
  const float* Wi    = (const float*)d_in[6];
  const float* bi_   = (const float*)d_in[7];
  const float* Wg    = (const float*)d_in[8];
  const float* bg_   = (const float*)d_in[9];
  const float* Wsi   = (const float*)d_in[10];
  const float* bsi   = (const float*)d_in[11];
  const float* Wsg   = (const float*)d_in[12];
  const float* bsg   = (const float*)d_in[13];
  const float* eWih  = (const float*)d_in[14];
  const float* eWhh  = (const float*)d_in[15];
  const float* ebih  = (const float*)d_in[16];
  const float* ebhh  = (const float*)d_in[17];
  const float* dWih0 = (const float*)d_in[18];
  const float* dWhh0 = (const float*)d_in[19];
  const float* dbih0 = (const float*)d_in[20];
  const float* dbhh0 = (const float*)d_in[21];
  const float* dWih1 = (const float*)d_in[22];
  const float* dWhh1 = (const float*)d_in[23];
  const float* dbih1 = (const float*)d_in[24];
  const float* dbhh1 = (const float*)d_in[25];
  const float* fcW   = (const float*)d_in[26];
  const float* fcb   = (const float*)d_in[27];

  bf16_t* ws = (bf16_t*)d_ws;

  prep_weights<<<512, 256, 0, stream>>>(Wf, Wo, Wi, Wg, eWih, eWhh, dWhh0, dWih1, dWhh1, ws);

  ea_seq2seq_main<<<16, 512, 0, stream>>>(
      x, s, bf_, bo_, bi_, bg_, Wsi, bsi, Wsg, bsg,
      ebih, ebhh, dWih0, dbih0, dbhh0, dbih1, dbhh1, fcW, fcb,
      ws + OFF_EA, ws + OFF_ENC, ws + OFF_D0, ws + OFF_D1,
      ws + OFF_SEQ, (float*)d_out);
}
